// AdaptiveRouter_15874199126031
// MI455X (gfx1250) — matrix operand order repaired and checked by reading, not compiled, not run on hardware
//
#include <hip/hip_runtime.h>
#include <math.h>

// ---------------------------------------------------------------- constants
#define BDIM   4096
#define HDIM   1024
#define HHALF  512
#define NVOC   50000
#define NPAD   50176     // 196 * 256
#define TOPK   64

typedef __attribute__((ext_vector_type(16))) __bf16          v16bf;
typedef __attribute__((ext_vector_type(8)))  float           v8f;
typedef __attribute__((ext_vector_type(8)))  unsigned short  ushort8;
typedef __attribute__((ext_vector_type(16))) unsigned short  ushort16;
typedef __attribute__((ext_vector_type(4)))  unsigned short  ushort4v;

// fp32 -> bf16 round-to-nearest-even
__device__ __forceinline__ unsigned short f2bf(float f) {
    unsigned int u = __float_as_uint(f);
    unsigned int r = u + 0x7FFFu + ((u >> 16) & 1u);
    return (unsigned short)(r >> 16);
}

__device__ __forceinline__ v16bf mkbf16(ushort8 lo, ushort8 hi) {
    ushort16 u;
#pragma unroll
    for (int i = 0; i < 8; ++i) { u[i] = lo[i]; u[i + 8] = hi[i]; }
    return __builtin_bit_cast(v16bf, u);
}

// A fragment: two 16B chunks at +0 and +16 elements
__device__ __forceinline__ v16bf loadA(const unsigned short* p) {
    return mkbf16(*(const ushort8*)p, *(const ushort8*)(p + 16));
}
// B fragment: two 16B chunks at +0 and +16 elements
__device__ __forceinline__ v16bf loadB(const unsigned short* p) {
    return mkbf16(*(const ushort8*)p, *(const ushort8*)(p + 16));
}

// ---------------------------------------------------------------- prep: add-vector + zero counts
__global__ void prep_kernel(const float* __restrict__ b2, const float* __restrict__ comp,
                            const float* __restrict__ ema, float* __restrict__ addv,
                            unsigned int* __restrict__ count) {
    int n = blockIdx.x * 256 + threadIdx.x;
    if (n < NVOC) {
        addv[n]  = b2[n] + 0.3f * comp[n] + 0.1f / (ema[n] + 1e-6f);
        count[n] = 0u;
    }
}

// ---------------------------------------------------------------- elementwise fp32 -> bf16
__global__ void convert_bf16_kernel(const float* __restrict__ in,
                                    unsigned short* __restrict__ out, int n) {
    int i = (blockIdx.x * 256 + threadIdx.x) * 4;
    if (i + 3 < n) {
        float4 v = *(const float4*)(in + i);
        ushort4v o; o.x = f2bf(v.x); o.y = f2bf(v.y); o.z = f2bf(v.z); o.w = f2bf(v.w);
        *(ushort4v*)(out + i) = o;
    }
}

// ---------------------------------------------------------------- transpose + convert
// in: [Kd, Nd] fp32 row-major.  out: [NpadRows, Kd] bf16, out[n*Kd + k] = in[k*Nd + n]
__global__ void transpose_conv_kernel(const float* __restrict__ in,
                                      unsigned short* __restrict__ out,
                                      int Kd, int Nd, int NpadRows) {
    __shared__ float t[32][33];
    int k0 = blockIdx.y * 32, n0 = blockIdx.x * 32;
#pragma unroll
    for (int i = threadIdx.y; i < 32; i += 8) {
        int k = k0 + i, n = n0 + threadIdx.x;
        t[i][threadIdx.x] = (k < Kd && n < Nd) ? in[(size_t)k * Nd + n] : 0.0f;
    }
    __syncthreads();
#pragma unroll
    for (int i = threadIdx.y; i < 32; i += 8) {
        int n = n0 + i, k = k0 + threadIdx.x;
        if (n < NpadRows && k < Kd)
            out[(size_t)n * Kd + k] = f2bf(t[threadIdx.x][i]);
    }
}

// ---------------------------------------------------------------- GEMM1 + bias + LayerNorm + GELU -> h_bf16
// z_bf16 [4096,1024] @ w1t [512,1024]^T -> 16 rows x 512 cols per block (8 waves x 64 cols)
__global__ __launch_bounds__(256) void gemm1_ln_gelu_kernel(
    const unsigned short* __restrict__ zbf, const unsigned short* __restrict__ w1t,
    const float* __restrict__ b1, const float* __restrict__ gamma,
    const float* __restrict__ beta, unsigned short* __restrict__ hbf) {
    __shared__ float tile[16 * 520];
    __shared__ float redS[16][16];
    __shared__ float redQ[16][16];
    __shared__ float muS[16], rsS[16];

    const int tid  = threadIdx.x;
    const int lane = tid & 31, wave = tid >> 5;
    const int m16 = lane & 15, kh = lane >> 4;
    const int rowBase = blockIdx.x * 16;

    v8f acc[4];
#pragma unroll
    for (int t = 0; t < 4; ++t) acc[t] = (v8f)0.0f;

    const unsigned short* aB = zbf + (size_t)(rowBase + m16) * HDIM + kh * 8;
    const unsigned short* bB = w1t + (size_t)(wave * 64 + m16) * HDIM + kh * 8;

    // ping-pong double-buffered fragments (no rotation copies)
    v16bf Af[2], Bf[2][4];
    auto loadStage = [&](int s, int kk) {
        Af[s] = loadA(aB + kk);
#pragma unroll
        for (int t = 0; t < 4; ++t) Bf[s][t] = loadB(bB + (size_t)t * 16 * HDIM + kk);
    };
    auto mmaStage = [&](int s) {
#pragma unroll
        for (int t = 0; t < 4; ++t)
            acc[t] = __builtin_amdgcn_wmma_f32_16x16x32_bf16(
                false, Af[s], false, Bf[s][t], (short)0, acc[t], false, false);
    };

    loadStage(0, 0);
    loadStage(1, 32);
    for (int k0 = 0; k0 < HDIM - 64; k0 += 64) {
        mmaStage(0); loadStage(0, k0 + 64);
        mmaStage(1); loadStage(1, k0 + 96);
    }
    mmaStage(0);
    mmaStage(1);

    // store accumulators (+bias) into LDS tile
#pragma unroll
    for (int t = 0; t < 4; ++t) {
        int col = wave * 64 + t * 16 + m16;
        float bias = b1[col];
#pragma unroll
        for (int i = 0; i < 8; ++i)
            tile[(i + 8 * kh) * 520 + col] = acc[t][i] + bias;
    }
    __syncthreads();

    // per-row mean / var partials
    {
        int r = tid >> 4, c = tid & 15;
        float s = 0.f, q = 0.f;
#pragma unroll 4
        for (int j = 0; j < 32; ++j) {
            float x = tile[r * 520 + c + 16 * j];
            s += x; q += x * x;
        }
        redS[r][c] = s; redQ[r][c] = q;
    }
    __syncthreads();
    if ((tid & 15) == 0) {
        int r = tid >> 4;
        float s = 0.f, q = 0.f;
#pragma unroll
        for (int c = 0; c < 16; ++c) { s += redS[r][c]; q += redQ[r][c]; }
        float mu  = s * (1.0f / 512.0f);
        float var = q * (1.0f / 512.0f) - mu * mu;
        muS[r] = mu; rsS[r] = rsqrtf(var + 1e-5f);
    }
    __syncthreads();

    // LN + exact GELU + bf16 convert, coalesced store
#pragma unroll 4
    for (int j = 0; j < 32; ++j) {
        int f = tid + 256 * j;
        int r = f >> 9, c = f & 511;
        float x  = tile[r * 520 + c];
        float xn = (x - muS[r]) * rsS[r] * gamma[c] + beta[c];
        float g  = 0.5f * xn * (1.0f + erff(xn * 0.70710678118654752f));
        hbf[(size_t)(rowBase + r) * HHALF + c] = f2bf(g);
    }
}

// ---------------------------------------------------------------- GEMM2 + score epilogue + mask zero-fill
// h_bf16 [4096,512] @ w2t [NPAD,512]^T.  Block: 64(M) x 256(N), 8 waves of 32x64.
__global__ __launch_bounds__(256) void gemm2_kernel(
    const unsigned short* __restrict__ hbf, const unsigned short* __restrict__ w2t,
    const float* __restrict__ addv, float* __restrict__ scores, float* __restrict__ mask) {
    const int lane = threadIdx.x & 31;
    const int wave = threadIdx.x >> 5;
    const int waveM = wave & 1, waveN = wave >> 1;
    const int rowBase = blockIdx.y * 64 + waveM * 32;
    const int colBase = blockIdx.x * 256 + waveN * 64;
    const int m16 = lane & 15, kh = lane >> 4;

    v8f acc[2][4];
#pragma unroll
    for (int s = 0; s < 2; ++s)
#pragma unroll
        for (int t = 0; t < 4; ++t) acc[s][t] = (v8f)0.0f;

    const unsigned short* aB0 = hbf + (size_t)(rowBase + m16) * HHALF + kh * 8;
    const unsigned short* aB1 = aB0 + (size_t)16 * HHALF;
    const unsigned short* bB  = w2t + (size_t)(colBase + m16) * HHALF + kh * 8;

    // ping-pong double-buffered fragments (no rotation copies)
    v16bf Af[2][2], Bf[2][4];
    auto loadStage = [&](int s, int kk) {
        Af[s][0] = loadA(aB0 + kk);
        Af[s][1] = loadA(aB1 + kk);
#pragma unroll
        for (int t = 0; t < 4; ++t) Bf[s][t] = loadB(bB + (size_t)t * 16 * HHALF + kk);
    };
    auto mmaStage = [&](int s) {
#pragma unroll
        for (int t = 0; t < 4; ++t) {
            acc[0][t] = __builtin_amdgcn_wmma_f32_16x16x32_bf16(
                false, Af[s][0], false, Bf[s][t], (short)0, acc[0][t], false, false);
            acc[1][t] = __builtin_amdgcn_wmma_f32_16x16x32_bf16(
                false, Af[s][1], false, Bf[s][t], (short)0, acc[1][t], false, false);
        }
    };

    loadStage(0, 0);
    loadStage(1, 32);
    for (int k0 = 0; k0 < HHALF - 64; k0 += 64) {
        mmaStage(0); loadStage(0, k0 + 64);
        mmaStage(1); loadStage(1, k0 + 96);
    }
    mmaStage(0);
    mmaStage(1);

    // epilogue: scores = acc + addv[col]; mask = 0  (non-temporal: keep w2t resident in L2)
#pragma unroll
    for (int s = 0; s < 2; ++s)
#pragma unroll
        for (int t = 0; t < 4; ++t) {
            int c = colBase + t * 16 + m16;
            if (c < NVOC) {
                float av = addv[c];
#pragma unroll
                for (int i = 0; i < 8; ++i) {
                    int r = rowBase + s * 16 + (kh * 8) + i;
                    size_t off = (size_t)r * NVOC + c;
                    __builtin_nontemporal_store(acc[s][t][i] + av, &scores[off]);
                    __builtin_nontemporal_store(0.0f, &mask[off]);
                }
            }
        }
}

// ---------------------------------------------------------------- top-64 per row (radix-select + exact argmax)
__global__ __launch_bounds__(256) void topk_kernel(
    const float* __restrict__ scores, float* __restrict__ mask,
    int* __restrict__ topIdx, unsigned int* __restrict__ count) {
    __shared__ unsigned int hist[4096];
    __shared__ unsigned int candKey[4096];
    __shared__ unsigned int candIdx[4096];
    __shared__ unsigned int gsum[256];
    __shared__ unsigned long long redBuf[256];
    __shared__ unsigned int sBin, nCand;

    const int row = blockIdx.x;
    const int tid = threadIdx.x;
    const float* s = scores + (size_t)row * NVOC;

    for (int i = tid; i < 4096; i += 256) hist[i] = 0u;
    if (tid == 0) nCand = 0u;
    __syncthreads();

    // pass 1: 12-bit histogram of order-preserving keys (non-temporal scan)
    for (int n = tid; n < NVOC; n += 256) {
        unsigned int u = __float_as_uint(__builtin_nontemporal_load(&s[n]));
        unsigned int k = (u & 0x80000000u) ? ~u : (u | 0x80000000u);
        atomicAdd(&hist[k >> 20], 1u);
    }
    __syncthreads();

    // find threshold bin (>= 64 elements in bins >= b*)
    {
        unsigned int g = 0;
#pragma unroll
        for (int j = 0; j < 16; ++j) g += hist[tid * 16 + j];
        gsum[tid] = g;
    }
    __syncthreads();
    if (tid == 0) {
        unsigned int acc = 0;
        int gi = 255;
        for (; gi > 0; --gi) { if (acc + gsum[gi] >= TOPK) break; acc += gsum[gi]; }
        int b = gi * 16 + 15;
        for (; b > gi * 16; --b) { if (acc + hist[b] >= TOPK) break; acc += hist[b]; }
        sBin = (unsigned int)b;
    }
    __syncthreads();
    const unsigned int bstar = sBin;

    // pass 2: collect all candidates with bin >= b*
    for (int n = tid; n < NVOC; n += 256) {
        unsigned int u = __float_as_uint(__builtin_nontemporal_load(&s[n]));
        unsigned int k = (u & 0x80000000u) ? ~u : (u | 0x80000000u);
        if ((k >> 20) >= bstar) {
            unsigned int p = atomicAdd(&nCand, 1u);
            if (p < 4096u) { candKey[p] = k; candIdx[p] = (unsigned int)n; }
        }
    }
    __syncthreads();

    int C = (nCand > 4096u) ? 4096 : (int)nCand;
    int need = (C < TOPK) ? C : TOPK;

    // iterative exact argmax: descending order, smaller index wins ties
    for (int t = 0; t < need; ++t) {
        unsigned long long best = 0ull;
        for (int i = tid; i < C; i += 256) {
            unsigned long long v =
                ((unsigned long long)candKey[i] << 32) | (unsigned int)(~candIdx[i]);
            if (v > best) best = v;
        }
        redBuf[tid] = best;
        __syncthreads();
        for (int off = 128; off > 0; off >>= 1) {
            if (tid < off) {
                if (redBuf[tid + off] > redBuf[tid]) redBuf[tid] = redBuf[tid + off];
            }
            __syncthreads();
        }
        unsigned long long chosen = redBuf[0];
        unsigned int ck = (unsigned int)(chosen >> 32);
        unsigned int ci = ~((unsigned int)chosen);
        if (tid == 0) {
            topIdx[(size_t)row * TOPK + t] = (int)ci;
            mask[(size_t)row * NVOC + ci]  = 1.0f;
            atomicAdd(&count[ci], 1u);
        }
        for (int i = tid; i < C; i += 256)
            if (candKey[i] == ck && candIdx[i] == ci) candKey[i] = 0u;
        __syncthreads();
    }
}

// ---------------------------------------------------------------- EMA update
__global__ void ema_kernel(const float* __restrict__ ema, const unsigned int* __restrict__ count,
                           float* __restrict__ outEma) {
    int n = blockIdx.x * 256 + threadIdx.x;
    if (n < NVOC) outEma[n] = 0.99f * ema[n] + 0.01f * (float)count[n];
}

// ---------------------------------------------------------------- launch
extern "C" void kernel_launch(void* const* d_in, const int* in_sizes, int n_in,
                              void* d_out, int out_size, void* d_ws, size_t ws_size,
                              hipStream_t stream) {
    const float* z     = (const float*)d_in[0];
    const float* w1    = (const float*)d_in[1];
    const float* b1    = (const float*)d_in[2];
    const float* gamma = (const float*)d_in[3];
    const float* beta  = (const float*)d_in[4];
    const float* w2    = (const float*)d_in[5];
    const float* b2    = (const float*)d_in[6];
    const float* comp  = (const float*)d_in[7];
    const float* ema   = (const float*)d_in[8];

    float* outF   = (float*)d_out;
    float* mask   = outF;                                             // [B, N]
    int*   topIdx = (int*)(outF + (size_t)BDIM * NVOC);               // [B, 64] int32
    float* scores = outF + (size_t)BDIM * NVOC + (size_t)BDIM * TOPK; // [B, N]
    float* newEma = scores + (size_t)BDIM * NVOC;                     // [N]

    // workspace layout (~65.5 MB)
    char* ws = (char*)d_ws;
    unsigned short* w2t  = (unsigned short*)(ws);                     // NPAD*512*2 = 51,380,224
    unsigned short* zbf  = (unsigned short*)(ws + 51380224);          // 8,388,608
    unsigned short* w1t  = (unsigned short*)(ws + 59768832);          // 1,048,576
    unsigned short* hbf  = (unsigned short*)(ws + 60817408);          // 4,194,304
    float*          addv = (float*)(ws + 65011712);                   // 200,704
    unsigned int*   cnt  = (unsigned int*)(ws + 65212416);            // 200,704

    prep_kernel<<<196, 256, 0, stream>>>(b2, comp, ema, addv, cnt);
    convert_bf16_kernel<<<(BDIM * HDIM / 4 + 255) / 256, 256, 0, stream>>>(z, zbf, BDIM * HDIM);
    transpose_conv_kernel<<<dim3(16, 32), dim3(32, 8), 0, stream>>>(w1, w1t, HDIM, HHALF, HHALF);
    transpose_conv_kernel<<<dim3(NPAD / 32, 16), dim3(32, 8), 0, stream>>>(w2, w2t, HHALF, NVOC, NPAD);
    gemm1_ln_gelu_kernel<<<BDIM / 16, 256, 0, stream>>>(zbf, w1t, b1, gamma, beta, hbf);
    gemm2_kernel<<<dim3(NPAD / 256, BDIM / 64), 256, 0, stream>>>(hbf, w2t, addv, scores, mask);
    topk_kernel<<<BDIM, 256, 0, stream>>>(scores, mask, topIdx, cnt);
    ema_kernel<<<196, 256, 0, stream>>>(ema, cnt, newEma);
}
